// GRU_69844758168130
// MI455X (gfx1250) — compile-verified, matrix operand order checked by reading, not run on hardware
//
#include <hip/hip_runtime.h>

#define DEVINL __device__ __forceinline__

typedef __attribute__((ext_vector_type(16))) __bf16      v16bf;
typedef __attribute__((ext_vector_type(8)))  float       v8f;
typedef __attribute__((ext_vector_type(4)))  unsigned int u32x4;

union FragU { v16bf v; u32x4 q[2]; };

constexpr int V_ = 32000, E_ = 512, H_ = 1024, L_ = 2, T_ = 64, B_ = 64;

DEVINL unsigned short f32_to_bf16(float f) {
  unsigned int u = __float_as_uint(f);
  return (unsigned short)((u + 0x7FFFu + ((u >> 16) & 1u)) >> 16);
}

// Load one 16x32 bf16 fragment (A layout; B uses same layout with row = out-channel).
// lane<16: row = row0+lane,    K chunks at k0+0  and k0+16
// lane>=16: row = row0+lane-16, K chunks at k0+8 and k0+24
DEVINL v16bf load_frag(const unsigned short* __restrict__ base, int ld, int row0,
                       int k0, int lane) {
  const int r  = row0 + (lane & 15);
  const int kb = k0 + ((lane >> 4) << 3);
  const unsigned short* p = base + (size_t)r * ld + kb;
  FragU f;
  f.q[0] = *(const u32x4*)(p);
  f.q[1] = *(const u32x4*)(p + 16);
  return f.v;
}

// Same fragment load from an LDS tile with leading dim 64 (emits ds_load_b128).
DEVINL v16bf load_frag_lds(const unsigned short* base, int row0, int k0, int lane) {
  const int r  = row0 + (lane & 15);
  const int kb = k0 + ((lane >> 4) << 3);
  const unsigned short* p = base + r * 64 + kb;
  FragU f;
  f.q[0] = *(const u32x4*)(p);
  f.q[1] = *(const u32x4*)(p + 16);
  return f.v;
}

DEVINL v8f wmma_bf16(v16bf a, v16bf b, v8f c) {
  return __builtin_amdgcn_wmma_f32_16x16x32_bf16(false, a, false, b,
                                                 (short)0, c, false, false);
}

// Async global->LDS copy of one 16-byte chunk (ASYNCcnt-tracked, per ISA 08 §4).
DEVINL void async_g2l_b128(const unsigned short* gsrc, const unsigned short* lds_dst) {
  unsigned lds_off = (unsigned)(unsigned long long)(const void*)lds_dst;
  asm volatile("global_load_async_to_lds_b128 %0, %1, off"
               :: "v"(lds_off), "v"(gsrc) : "memory");
}
DEVINL void wait_asynccnt_le2() { asm volatile("s_wait_asynccnt 0x2" ::: "memory"); }
DEVINL void wait_asynccnt_0()   { asm volatile("s_wait_asynccnt 0x0" ::: "memory"); }

DEVINL float sigmoidf_(float x) { return 1.0f / (1.0f + __expf(-x)); }

// ---------------------------------------------------------------- utilities
__global__ void cvt_f32_bf16_kernel(const float* __restrict__ src,
                                    unsigned short* __restrict__ dst, int n) {
  int i = blockIdx.x * blockDim.x + threadIdx.x;
  int stride = gridDim.x * blockDim.x;
  for (; i < n; i += stride) dst[i] = f32_to_bf16(src[i]);
}

__global__ void gather_emb_kernel(const int* __restrict__ toks,
                                  const float* __restrict__ emb,
                                  unsigned short* __restrict__ X) {
  const int row = blockIdx.x;                 // 0..T*B-1
  const int tok = toks[row];
  const float* src = emb + (size_t)tok * E_;
  unsigned short* dst = X + (size_t)row * E_;
  for (int e = threadIdx.x; e < E_; e += blockDim.x) dst[e] = f32_to_bf16(src[e]);
}

__global__ void init_hidden_kernel(const float* __restrict__ h0,
                                   float* __restrict__ hF,
                                   unsigned short* __restrict__ hbf, int n) {
  int i = blockIdx.x * blockDim.x + threadIdx.x;
  int stride = gridDim.x * blockDim.x;
  for (; i < n; i += stride) { float v = h0[i]; hF[i] = v; hbf[i] = f32_to_bf16(v); }
}

__global__ void copy_final_kernel(const float* __restrict__ hF,
                                  float* __restrict__ dst, int n) {
  int i = blockIdx.x * blockDim.x + threadIdx.x;
  int stride = gridDim.x * blockDim.x;
  for (; i < n; i += stride) dst[i] = hF[i];
}

// ------------------------------------------------------- full GRU scan (1 launch)
// grid = 4 blocks (16 batch rows each), block = 512 threads = 16 waves,
// each wave owns 4 N-tiles of 16 (covers H=1024).
__global__ void __launch_bounds__(512, 1)
gru_scan_kernel(const unsigned short* __restrict__ Xemb,   // [T*B, E]
                unsigned short* __restrict__ hbf,          // [L, B, H] bf16
                float* __restrict__ hF,                    // [L, B, H] f32
                unsigned short* __restrict__ rhbuf,        // [B, H] bf16
                float* __restrict__ zbuf,                  // [B, H]
                float* __restrict__ cbuf,                  // [B, H]
                unsigned short* __restrict__ Xcoll,        // [T*B, H] bf16
                const unsigned short* __restrict__ wIr0,
                const unsigned short* __restrict__ wIz0,
                const unsigned short* __restrict__ wIh0,
                const unsigned short* __restrict__ wIr1,
                const unsigned short* __restrict__ wIz1,
                const unsigned short* __restrict__ wIh1,
                const unsigned short* __restrict__ wHr,
                const unsigned short* __restrict__ wHz,
                const unsigned short* __restrict__ wHh,
                const float* __restrict__ b_hr,
                const float* __restrict__ b_hz,
                const float* __restrict__ b_hh) {
  const int tid    = threadIdx.x;
  const int lane   = tid & 31;
  const int wave   = tid >> 5;
  const int m0     = blockIdx.x * 16;           // this block's batch rows
  const int colN   = lane & 15;
  const int rowOff = (lane >> 4) << 3;

  for (int t = 0; t < T_; ++t) {
    for (int l = 0; l < L_; ++l) {
      const int Kx = (l == 0) ? E_ : H_;
      const unsigned short* xbf = (l == 0) ? (Xemb + (size_t)t * B_ * E_) : hbf;
      const unsigned short* Wir = (l == 0) ? wIr0 : wIr1;
      const unsigned short* Wiz = (l == 0) ? wIz0 : wIz1;
      const unsigned short* Wih = (l == 0) ? wIh0 : wIh1;
      const unsigned short* Whr  = wHr + (size_t)l * H_ * H_;
      const unsigned short* Whz  = wHz + (size_t)l * H_ * H_;
      const unsigned short* Whh_ = wHh + (size_t)l * H_ * H_;
      const unsigned short* hbl  = hbf + (size_t)l * B_ * H_;
      unsigned short* hbl_w      = hbf + (size_t)l * B_ * H_;
      float* hFl                 = hF  + (size_t)l * B_ * H_;
      const float* bhr = b_hr + l * H_;
      const float* bhz = b_hz + l * H_;
      const float* bhh = b_hh + l * H_;

      // ---- phase A: r, z gates and input part of c -------------------------
      for (int j = 0; j < 4; ++j) {
        const int n0 = (wave * 4 + j) * 16;
        v8f racc = {}, zacc = {}, cacc = {};
        for (int k = 0; k < Kx; k += 32) {
          v16bf ax = load_frag(xbf, Kx, m0, k, lane);
          racc = wmma_bf16(ax, load_frag(Wir, Kx, n0, k, lane), racc);
          zacc = wmma_bf16(ax, load_frag(Wiz, Kx, n0, k, lane), zacc);
          cacc = wmma_bf16(ax, load_frag(Wih, Kx, n0, k, lane), cacc);
        }
        for (int k = 0; k < H_; k += 32) {
          v16bf ah = load_frag(hbl, H_, m0, k, lane);
          racc = wmma_bf16(ah, load_frag(Whr, H_, n0, k, lane), racc);
          zacc = wmma_bf16(ah, load_frag(Whz, H_, n0, k, lane), zacc);
        }
        const int n = n0 + colN;
        const float br = bhr[n], bz = bhz[n];
#pragma unroll
        for (int i = 0; i < 8; ++i) {
          const int row = m0 + rowOff + i;
          const float r  = sigmoidf_(racc[i] + br);
          const float z  = sigmoidf_(zacc[i] + bz);
          const float hv = hFl[row * H_ + n];
          rhbuf[row * H_ + n] = f32_to_bf16(r * hv);
          zbuf[row * H_ + n]  = z;
          cbuf[row * H_ + n]  = cacc[i];
        }
      }
      __threadfence_block();
      __syncthreads();

      // ---- phase B: c = tanh(cpart + (r*h)@Whh^T + b), h' -----------------
      for (int j = 0; j < 4; ++j) {
        const int n0 = (wave * 4 + j) * 16;
        v8f acc = {};
        for (int k = 0; k < H_; k += 32) {
          acc = wmma_bf16(load_frag(rhbuf, H_, m0, k, lane),
                          load_frag(Whh_, H_, n0, k, lane), acc);
        }
        const int n = n0 + colN;
        const float bh = bhh[n];
#pragma unroll
        for (int i = 0; i < 8; ++i) {
          const int row = m0 + rowOff + i;
          const float c  = tanhf(cbuf[row * H_ + n] + acc[i] + bh);
          const float z  = zbuf[row * H_ + n];
          const float hv = hFl[row * H_ + n];
          const float hn = tanhf((1.0f - z) * hv + z * c);
          hFl[row * H_ + n] = hn;
          const unsigned short hb = f32_to_bf16(hn);
          hbl_w[row * H_ + n] = hb;
          if (l == 1) Xcoll[((size_t)t * B_ + row) * H_ + n] = hb;
        }
      }
      __threadfence_block();
      __syncthreads();
    }
  }
}

// ---------------------------------------------- logits GEMM  [4096,1024]x[1024,32000]
// Block: 8 waves share one 64-row M-strip of X (staged in LDS via async copies,
// double buffered, 64 K-columns per chunk); each wave owns one 16-col N-tile and
// streams its W_out rows from global.
__global__ void __launch_bounds__(256, 2)
logits_kernel(const unsigned short* __restrict__ X,    // [T*B, H]
              const unsigned short* __restrict__ Wo,   // [V, H]
              const float* __restrict__ bo,            // [V]
              float* __restrict__ out) {               // [T*B, V]
  __shared__ unsigned short Atile[2][64 * 64];         // 2 x 8KB ping-pong

  const int tid    = threadIdx.x;
  const int lane   = tid & 31;
  const int wave   = tid >> 5;
  const int mstrip = blockIdx.x / 250;           // 0..63
  const int ngroup = blockIdx.x % 250;           // 0..249
  const int n0     = (ngroup * 8 + wave) * 16;   // 0..31984
  const int mb     = mstrip * 64;

  const unsigned short* wrow = Wo + (size_t)(n0 + (lane & 15)) * H_;

  // cooperative async copy of a 64x64 bf16 A chunk: 512 16B chunks / 256 threads
  auto issue_chunk = [&](int buf, int kc) {
#pragma unroll
    for (int h = 0; h < 2; ++h) {
      const int c   = tid + h * 256;
      const int row = c >> 3;
      const int col = (c & 7) * 8;
      async_g2l_b128(X + (size_t)(mb + row) * H_ + (kc + col),
                     &Atile[buf][row * 64 + col]);
    }
  };

  v8f acc0 = {}, acc1 = {}, acc2 = {}, acc3 = {};
  issue_chunk(0, 0);
  int buf = 0;
  for (int kc = 0; kc < H_; kc += 64) {
    const bool more = (kc + 64) < H_;
    if (more) issue_chunk(buf ^ 1, kc + 64);     // prefetch next chunk
    if (more) wait_asynccnt_le2();               // current chunk landed (in-order)
    else      wait_asynccnt_0();
    __syncthreads();
#pragma unroll
    for (int ks = 0; ks < 64; ks += 32) {
      __builtin_prefetch(wrow + kc + ks + 128, 0, 0);   // stream W_out
      v16bf bf = load_frag(Wo, H_, n0, kc + ks, lane);
      acc0 = wmma_bf16(load_frag_lds(&Atile[buf][0], 0,  ks, lane), bf, acc0);
      acc1 = wmma_bf16(load_frag_lds(&Atile[buf][0], 16, ks, lane), bf, acc1);
      acc2 = wmma_bf16(load_frag_lds(&Atile[buf][0], 32, ks, lane), bf, acc2);
      acc3 = wmma_bf16(load_frag_lds(&Atile[buf][0], 48, ks, lane), bf, acc3);
    }
    __syncthreads();                             // done reading buf before overwrite
    buf ^= 1;
  }

  const int n      = n0 + (lane & 15);
  const float bn   = bo[n];
  const int rowOff = (lane >> 4) << 3;
  v8f accs[4] = {acc0, acc1, acc2, acc3};
#pragma unroll
  for (int i2 = 0; i2 < 4; ++i2) {
#pragma unroll
    for (int i = 0; i < 8; ++i) {
      const int row = mb + i2 * 16 + rowOff + i;
      out[(size_t)row * V_ + n] = accs[i2][i] + bn;
    }
  }
}

// --------------------------------------------------------------------- launch
extern "C" void kernel_launch(void* const* d_in, const int* in_sizes, int n_in,
                              void* d_out, int out_size, void* d_ws, size_t ws_size,
                              hipStream_t stream) {
  (void)in_sizes; (void)n_in; (void)out_size; (void)ws_size;

  const int*   inputs = (const int*)  d_in[0];
  const float* hidden = (const float*)d_in[1];
  const float* emb    = (const float*)d_in[2];
  const float* fW_ir0 = (const float*)d_in[3];
  const float* fW_iz0 = (const float*)d_in[4];
  const float* fW_ih0 = (const float*)d_in[5];
  const float* fW_ir1 = (const float*)d_in[6];
  const float* fW_iz1 = (const float*)d_in[7];
  const float* fW_ih1 = (const float*)d_in[8];
  const float* fW_hr  = (const float*)d_in[9];
  const float* b_hr   = (const float*)d_in[10];
  const float* fW_hz  = (const float*)d_in[11];
  const float* b_hz   = (const float*)d_in[12];
  const float* fW_hh  = (const float*)d_in[13];
  const float* b_hh   = (const float*)d_in[14];
  const float* fW_out = (const float*)d_in[15];
  const float* b_out  = (const float*)d_in[16];
  float* out = (float*)d_out;

  char* ws = (char*)d_ws;
  size_t off = 0;
  auto alloc = [&](size_t bytes) -> void* {
    void* p = ws + off;
    off = (off + bytes + 255) & ~(size_t)255;
    return p;
  };

  unsigned short* wIr0 = (unsigned short*)alloc((size_t)H_ * E_ * 2);
  unsigned short* wIz0 = (unsigned short*)alloc((size_t)H_ * E_ * 2);
  unsigned short* wIh0 = (unsigned short*)alloc((size_t)H_ * E_ * 2);
  unsigned short* wIr1 = (unsigned short*)alloc((size_t)H_ * H_ * 2);
  unsigned short* wIz1 = (unsigned short*)alloc((size_t)H_ * H_ * 2);
  unsigned short* wIh1 = (unsigned short*)alloc((size_t)H_ * H_ * 2);
  unsigned short* wHr  = (unsigned short*)alloc((size_t)L_ * H_ * H_ * 2);
  unsigned short* wHz  = (unsigned short*)alloc((size_t)L_ * H_ * H_ * 2);
  unsigned short* wHh  = (unsigned short*)alloc((size_t)L_ * H_ * H_ * 2);
  unsigned short* wOut = (unsigned short*)alloc((size_t)V_ * H_ * 2);
  unsigned short* Xemb = (unsigned short*)alloc((size_t)T_ * B_ * E_ * 2);
  unsigned short* Xcol = (unsigned short*)alloc((size_t)T_ * B_ * H_ * 2);
  float*          hF   = (float*)         alloc((size_t)L_ * B_ * H_ * 4);
  unsigned short* hbf  = (unsigned short*)alloc((size_t)L_ * B_ * H_ * 2);
  unsigned short* rh   = (unsigned short*)alloc((size_t)B_ * H_ * 2);
  float*          zb   = (float*)         alloc((size_t)B_ * H_ * 4);
  float*          cb   = (float*)         alloc((size_t)B_ * H_ * 4);

  auto cvt = [&](const float* s, unsigned short* d, int n) {
    int grid = (n + 1023) / 1024;
    if (grid > 8192) grid = 8192;
    cvt_f32_bf16_kernel<<<dim3(grid), dim3(256), 0, stream>>>(s, d, n);
  };
  cvt(fW_ir0, wIr0, H_ * E_);
  cvt(fW_iz0, wIz0, H_ * E_);
  cvt(fW_ih0, wIh0, H_ * E_);
  cvt(fW_ir1, wIr1, H_ * H_);
  cvt(fW_iz1, wIz1, H_ * H_);
  cvt(fW_ih1, wIh1, H_ * H_);
  cvt(fW_hr,  wHr,  L_ * H_ * H_);
  cvt(fW_hz,  wHz,  L_ * H_ * H_);
  cvt(fW_hh,  wHh,  L_ * H_ * H_);
  cvt(fW_out, wOut, V_ * H_);

  gather_emb_kernel<<<dim3(T_ * B_), dim3(128), 0, stream>>>(inputs, emb, Xemb);
  init_hidden_kernel<<<dim3((L_ * B_ * H_ + 255) / 256), dim3(256), 0, stream>>>(
      hidden, hF, hbf, L_ * B_ * H_);

  gru_scan_kernel<<<dim3(B_ / 16), dim3(512), 0, stream>>>(
      Xemb, hbf, hF, rh, zb, cb, Xcol,
      wIr0, wIz0, wIh0, wIr1, wIz1, wIh1, wHr, wHz, wHh,
      b_hr, b_hz, b_hh);

  logits_kernel<<<dim3(64 * 250), dim3(256), 0, stream>>>(Xcol, wOut, b_out, out);

  copy_final_kernel<<<dim3((L_ * B_ * H_ + 255) / 256), dim3(256), 0, stream>>>(
      hF, out + (size_t)T_ * B_ * V_, L_ * B_ * H_);
}
